// SubGraph_30537217474769
// MI455X (gfx1250) — compile-verified, matrix operand order checked by reading, not run on hardware
//
#include <hip/hip_runtime.h>

// Fused 3-block MLP + masked max-pool for MI455X (gfx1250, wave32, WMMA).
//
// Mapping: one wave per (b,p) token. Per token: V=32 rows.
//   block1: [32x16] @ [16x64]   (K zero-padded to 32)
//   block2: [32x128] @ [128x64] (input = concat(out1, pool1))
//   block3: [32x128] @ [128x64] -> pooled 64-vector = output
// All GEMMs via v_wmma_f32_16x16x32_f16: M-tiles=2, N-tiles=4, K-steps=K/32.
// Activations round-trip through padded LDS; the "rep" half of the concat is
// never materialized per-row — those A fragments broadcast from a 64-entry
// pool buffer in LDS.

typedef __attribute__((ext_vector_type(16))) _Float16 v16h;
typedef __attribute__((ext_vector_type(8)))  _Float16 v8h;
typedef __attribute__((ext_vector_type(8)))  float    v8f;
typedef __attribute__((ext_vector_type(4)))  float    v4f;

namespace {
constexpr int kWaves = 8;    // waves per workgroup
constexpr int kV     = 32;   // rows per token
constexpr int kF     = 16;   // block-1 input features
constexpr int kH     = 64;   // hidden size
constexpr int kXS    = 72;   // LDS activation row stride (halves): 64 + 8 pad
constexpr int kWS    = 136;  // LDS W2/W3 row stride (halves): 128 + 8 pad
constexpr int kW1S   = 40;   // LDS W1 row stride (halves): 32 + 8 pad
}

__device__ __forceinline__ v8f wmma16(v16h a, v16h b, v8f c) {
  // v_wmma_f32_16x16x32_f16  (neg_a, A, neg_b, B, c_mod, C, reuse_a, reuse_b)
  return __builtin_amdgcn_wmma_f32_16x16x32_f16(false, a, false, b,
                                                (short)0, c, false, false);
}

// Build a 16x32 f16 fragment from row-major LDS. Caller passes per-lane base:
//   base = rowbase + (lane&15)*stride + (lane>=16 ? 8 : 0)
// Lane holds K{+0..7} (VGPR0-3) and K{+16..23} (VGPR4-7) -> two ds_load_b128.
__device__ __forceinline__ v16h frag_ld(const _Float16* p) {
  v8h lo = *(const v8h*)(p);
  v8h hi = *(const v8h*)(p + 16);
  v16h r;
  #pragma unroll
  for (int i = 0; i < 8; ++i) { r[i] = lo[i]; r[i + 8] = hi[i]; }
  return r;
}

__global__ __launch_bounds__(kWaves * 32)
void fused_mlp_maxpool(const float* __restrict__ xg,
                       const unsigned char* __restrict__ maskg,
                       const float* __restrict__ W1g, const float* __restrict__ b1g,
                       const float* __restrict__ W2g, const float* __restrict__ b2g,
                       const float* __restrict__ W3g, const float* __restrict__ b3g,
                       float* __restrict__ outg, int ntok)
{
  __shared__ _Float16 sW1[kH * kW1S];           //  5.0 KB
  __shared__ _Float16 sW2[kH * kWS];            // 17.0 KB
  __shared__ _Float16 sW3[kH * kWS];            // 17.0 KB
  __shared__ _Float16 sX[kWaves][kV * kXS];     // 36.0 KB (per-wave slices)
  __shared__ _Float16 sP[kWaves][kH];           //  1.0 KB

  const int tid = threadIdx.x;

  // ---- one-time weight staging: f32 global -> f16 LDS, padded strides ----
  for (int i = tid; i < kH * kW1S; i += kWaves * 32) {
    int r = i / kW1S, c = i - r * kW1S;
    sW1[i] = (_Float16)((c < kF) ? W1g[r * kF + c] : 0.0f);  // zero-pad K 16..31
  }
  for (int i = tid; i < kH * kWS; i += kWaves * 32) {
    int r = i / kWS, c = i - r * kWS;
    float w2 = (c < 2 * kH) ? W2g[r * 2 * kH + c] : 0.0f;
    float w3 = (c < 2 * kH) ? W3g[r * 2 * kH + c] : 0.0f;
    sW2[i] = (_Float16)w2;
    sW3[i] = (_Float16)w3;
  }
  __syncthreads();

  const int wid  = tid >> 5;
  const int lane = tid & 31;
  const int l15  = lane & 15;
  const int hi8  = (lane >> 4) * 8;   // 0 for lanes 0-15, 8 for lanes 16-31
  _Float16* xb = &sX[wid][0];
  _Float16* pb = &sP[wid][0];

  // per-lane bias for column n*16 + l15, per block
  float bias[3][4];
  #pragma unroll
  for (int n = 0; n < 4; ++n) {
    bias[0][n] = b1g[n * 16 + l15];
    bias[1][n] = b2g[n * 16 + l15];
    bias[2][n] = b3g[n * 16 + l15];
  }

  const v8f zero8 = {0.f, 0.f, 0.f, 0.f, 0.f, 0.f, 0.f, 0.f};
  const int gw = blockIdx.x * kWaves + wid;
  const int nw = gridDim.x * kWaves;

  for (int t = gw; t < ntok; t += nw) {
    // ---- stage input tile: 32 rows x 16 f32 -> f16 LDS, zero-pad K 16..31 ----
    {
      const float* g = xg + (size_t)t * (kV * kF) + lane * kF;
      v4f x0 = *(const v4f*)(g + 0);
      v4f x1 = *(const v4f*)(g + 4);
      v4f x2 = *(const v4f*)(g + 8);
      v4f x3 = *(const v4f*)(g + 12);
      v8h h0, h1, z;
      #pragma unroll
      for (int i = 0; i < 4; ++i) {
        h0[i]     = (_Float16)x0[i];  h0[i + 4] = (_Float16)x1[i];
        h1[i]     = (_Float16)x2[i];  h1[i + 4] = (_Float16)x3[i];
        z[i]      = (_Float16)0.0f;   z[i + 4]  = (_Float16)0.0f;
      }
      _Float16* row = xb + lane * kXS;
      *(v8h*)(row +  0) = h0;
      *(v8h*)(row +  8) = h1;
      *(v8h*)(row + 16) = z;
      *(v8h*)(row + 24) = z;
    }

    const unsigned char mb = maskg[(size_t)t * kV + lane];
    const unsigned int mask32 = (unsigned int)__ballot(mb != 0);
    const bool anyv = (mask32 != 0u);

    float pool[4];  // pooled value for column n*16+l15 (valid in both halves)

    // epilogue: bias + leaky-relu + mask + masked-max pool (+optional store)
    auto post = [&](v8f (&acc)[2][4], const float* bn, bool store_out) {
      #pragma unroll
      for (int n = 0; n < 4; ++n) {
        float pm = -3.402823466e38f;
        #pragma unroll
        for (int m = 0; m < 2; ++m) {
          const unsigned int mr = mask32 >> (m * 16 + hi8);
          #pragma unroll
          for (int r = 0; r < 8; ++r) {
            float h = acc[m][n][r] + bn[n];
            float v = fmaxf(h, 0.0f) + 0.01f * fminf(h, 0.0f);  // leaky relu
            bool mk = (mr >> r) & 1u;
            float o = mk ? v : 0.0f;
            if (store_out)
              xb[(m * 16 + hi8 + r) * kXS + (n * 16 + l15)] = (_Float16)o;
            pm = fmaxf(pm, mk ? v : -3.402823466e38f);
          }
        }
        pm = fmaxf(pm, __shfl_xor(pm, 16, 32));  // combine lane halves
        pool[n] = anyv ? pm : 0.0f;
      }
      if (store_out && lane < 16) {
        #pragma unroll
        for (int n = 0; n < 4; ++n) pb[n * 16 + l15] = (_Float16)pool[n];
      }
    };

    // ================= block 1: K = 16 (padded to 32) =================
    {
      v16h a0 = frag_ld(xb + (0 * 16 + l15) * kXS + hi8);
      v16h a1 = frag_ld(xb + (1 * 16 + l15) * kXS + hi8);
      v8f acc[2][4];
      #pragma unroll
      for (int m = 0; m < 2; ++m)
        #pragma unroll
        for (int n = 0; n < 4; ++n) acc[m][n] = zero8;
      #pragma unroll
      for (int n = 0; n < 4; ++n) {
        v16h bf = frag_ld(sW1 + (n * 16 + l15) * kW1S + hi8);
        acc[0][n] = wmma16(a0, bf, acc[0][n]);
        acc[1][n] = wmma16(a1, bf, acc[1][n]);
      }
      post(acc, bias[0], true);
    }

    // ================= blocks 2 & 3: K = 128 =================
    #pragma unroll
    for (int blk = 1; blk < 3; ++blk) {
      const _Float16* sW = (blk == 1) ? sW2 : sW3;
      v16h a[2][4];
      #pragma unroll
      for (int m = 0; m < 2; ++m)
        #pragma unroll
        for (int k = 0; k < 2; ++k)  // "out" half from activation buffer
          a[m][k] = frag_ld(xb + (m * 16 + l15) * kXS + k * 32 + hi8);
      #pragma unroll
      for (int k = 2; k < 4; ++k) {  // "rep" half: broadcast from pool buffer
        v16h ap = frag_ld(pb + (k - 2) * 32 + hi8);
        a[0][k] = ap;
        a[1][k] = ap;
      }
      v8f acc[2][4];
      #pragma unroll
      for (int m = 0; m < 2; ++m)
        #pragma unroll
        for (int n = 0; n < 4; ++n) acc[m][n] = zero8;
      #pragma unroll
      for (int n = 0; n < 4; ++n) {
        #pragma unroll
        for (int k = 0; k < 4; ++k) {
          v16h bf = frag_ld(sW + (n * 16 + l15) * kWS + k * 32 + hi8);
          acc[0][n] = wmma16(a[0][k], bf, acc[0][n]);
          acc[1][n] = wmma16(a[1][k], bf, acc[1][n]);
        }
      }
      post(acc, bias[blk], blk == 1);
    }

    // output: pooled 64-vector of block 3 (already zeroed when !anyv)
    if (lane < 16) {
      #pragma unroll
      for (int n = 0; n < 4; ++n)
        outg[(size_t)t * kH + n * 16 + lane] = pool[n];
    }
  }
}

extern "C" void kernel_launch(void* const* d_in, const int* in_sizes, int n_in,
                              void* d_out, int out_size, void* d_ws, size_t ws_size,
                              hipStream_t stream) {
  const float*         xg = (const float*)d_in[0];
  const unsigned char* mg = (const unsigned char*)d_in[1];  // jnp.bool_: 1 byte
  const float*         W1 = (const float*)d_in[2];
  const float*         b1 = (const float*)d_in[3];
  const float*         W2 = (const float*)d_in[4];
  const float*         b2 = (const float*)d_in[5];
  const float*         W3 = (const float*)d_in[6];
  const float*         b3 = (const float*)d_in[7];
  float* out = (float*)d_out;

  const int ntok = in_sizes[1] / kV;  // B*P (mask flat count / V)
  if (ntok <= 0) return;
  int blocks = (ntok + kWaves - 1) / kWaves;
  if (blocks > 512) blocks = 512;     // ~4 tokens per wave at full size

  fused_mlp_maxpool<<<blocks, kWaves * 32, 0, stream>>>(
      xg, mg, W1, b1, W2, b2, W3, b3, out, ntok);
}
